// PredRNN_Model_61503931678832
// MI455X (gfx1250) — compile-verified, matrix operand order checked by reading, not run on hardware
//
#include <hip/hip_runtime.h>

// ---------------------------------------------------------------------------
// PredRNN (SpatioTemporalLSTM, L=4, T=19, B=8, 32x32, NH=128, K=5) for gfx1250.
// All convolutions are implicit GEMMs executed with v_wmma_f32_16x16x32_bf16.
// Gate convs (x/h/m) are fused into one GEMM via packed concatenated weights.
// Fragments are loaded as two ds_load_b128 per operand (ISA 7.12.2 layout).
// ---------------------------------------------------------------------------

#define L_LAYERS 4
#define NH 128
#define FC 16
#define HP 32
#define WP 32
#define HW 1024            // 32*32
#define BATCH 8
#define T_TOTAL 19
#define T_PREV 10
#define NPIX (BATCH * HW)  // 8192
#define GATES7 (7 * NH)    // 896
#define LDSPITCH 40        // shorts per row: 80B -> conflict-free b128 reads

typedef __attribute__((ext_vector_type(16))) __bf16 v16bf;
typedef __attribute__((ext_vector_type(8)))  float  v8f;

union FragAB { v16bf v; uint4 q[2]; unsigned short s[16]; };

__device__ __forceinline__ unsigned short f2bf(float f) {
    union { float f; unsigned u; } x; x.f = f;
    unsigned u = x.u;
    u += 0x7FFFu + ((u >> 16) & 1u);     // round to nearest even
    return (unsigned short)(u >> 16);
}
__device__ __forceinline__ float bf2f(unsigned short h) {
    union { unsigned u; float f; } x; x.u = ((unsigned)h) << 16;
    return x.f;
}
__device__ __forceinline__ float sigm(float x) { return 1.0f / (1.0f + __expf(-x)); }

// ---------------------------------------------------------------------------
// Generic implicit-GEMM conv kernel (SAME pad for taps==25, plain for taps==1)
//   W : [M, Kpad] bf16 (rows < M valid; Kpad % 32 == 0; K >= Ktot zero-padded)
//   X : [B, Cin, 32, 32] bf16
//   Y : [B, M, 1024] fp32
// Block: 256 threads (8 waves). Tile: 128(M) x 128(pixels), K-step 32.
// Wave (4x2 grid): 32(M) x 64(N) = 2x4 WMMA 16x16 accumulators.
// Both LDS tiles are [row][k] so every fragment is two 16B ds_load_b128:
//   shorts [half*8 .. +8) = K(0..7 | 8..15), [16+half*8 ..) = K(16..23 | 24..31)
// ---------------------------------------------------------------------------
__global__ __launch_bounds__(256)
void conv_gemm_wmma(const unsigned short* __restrict__ W,
                    const unsigned short* __restrict__ X,
                    float* __restrict__ Y,
                    int M, int Cin, int Ktot, int Kpad, int taps)
{
    __shared__ __align__(16) unsigned short ldsA[128][LDSPITCH];  // [cout][k]
    __shared__ __align__(16) unsigned short ldsB[128][LDSPITCH];  // [pixel][k]

    const int tid   = threadIdx.x;
    const int lane  = tid & 31;
    const int wave  = tid >> 5;
    const int waveM = wave & 3;   // 32-row chunk
    const int waveN = wave >> 2;  // 64-col chunk

    const int m0    = blockIdx.x * 128;
    const int p0    = blockIdx.y * 128;     // global pixel base (within one image)
    const int bimg  = p0 / HW;
    const int poff0 = p0 % HW;

    v8f acc[2][4];
    #pragma unroll
    for (int i = 0; i < 2; ++i)
        #pragma unroll
        for (int j = 0; j < 4; ++j)
            acc[i][j] = (v8f){0.f,0.f,0.f,0.f,0.f,0.f,0.f,0.f};

    const int half = lane >> 4;
    const int l16  = lane & 15;

    // per-thread B-gather coordinates: one (pixel, k-half) pair per thread
    const int bpix = tid & 127;
    const int bkh  = (tid >> 7) << 4;           // 0 or 16 (shorts)
    const int bpp  = poff0 + bpix;
    const int bpy  = bpp >> 5;
    const int bpx  = bpp & 31;

    for (int k0 = 0; k0 < Kpad; k0 += 32) {
        // ---- A tile: 128x32 bf16 weights, 16B vector loads (2 chunks/thread)
        #pragma unroll
        for (int c = tid; c < 512; c += 256) {
            const int row = c >> 2;
            const int col = (c & 3) * 8;
            uint4 v = make_uint4(0u, 0u, 0u, 0u);
            const int gm = m0 + row;
            if (gm < M)
                v = *(const uint4*)(W + (size_t)gm * Kpad + k0 + col);
            *(uint4*)&ldsA[row][col] = v;
        }
        // prefetch next weight tile into caches (global_prefetch_b8)
        if (k0 + 32 < Kpad && (m0 + (tid >> 1)) < M)
            __builtin_prefetch(W + (size_t)(m0 + (tid >> 1)) * Kpad + k0 + 32, 0, 1);

        // ---- B tile: im2col gather, 16 consecutive k per thread,
        //      incremental (ci, tap) decomposition (no per-element div),
        //      written as two ds_store_b128
        {
            union { unsigned short s[16]; uint4 q[2]; } tb;
            int kg = k0 + bkh;
            int ci, r;
            if (taps == 25) { ci = kg / 25; r = kg - ci * 25; }
            else            { ci = kg;     r = 0; }
            #pragma unroll
            for (int u = 0; u < 16; ++u) {
                unsigned short val = 0;
                if (kg + u < Ktot) {
                    int dy = 0, dx = 0;
                    if (taps == 25) {
                        const int ry = r / 5;
                        dy = ry - 2;
                        dx = r - ry * 5 - 2;
                    }
                    const int yy = bpy + dy;
                    const int xx = bpx + dx;
                    if (yy >= 0 && yy < HP && xx >= 0 && xx < WP)
                        val = X[((size_t)bimg * Cin + ci) * HW + yy * WP + xx];
                }
                tb.s[u] = val;
                if (taps == 25) { if (++r == 25) { r = 0; ++ci; } }
                else            { ++ci; }
            }
            *(uint4*)&ldsB[bpix][bkh]     = tb.q[0];
            *(uint4*)&ldsB[bpix][bkh + 8] = tb.q[1];
        }
        __syncthreads();

        // ---- fragments: two b128 LDS reads per operand
        FragAB aF[2], bF[4];
        #pragma unroll
        for (int mi = 0; mi < 2; ++mi) {
            const int mrow = waveM * 32 + mi * 16 + l16;   // A: M = lane%16
            aF[mi].q[0] = *(const uint4*)&ldsA[mrow][half * 8];
            aF[mi].q[1] = *(const uint4*)&ldsA[mrow][16 + half * 8];
        }
        #pragma unroll
        for (int ni = 0; ni < 4; ++ni) {
            const int nrow = waveN * 64 + ni * 16 + l16;   // B: N = lane%16
            bF[ni].q[0] = *(const uint4*)&ldsB[nrow][half * 8];
            bF[ni].q[1] = *(const uint4*)&ldsB[nrow][16 + half * 8];
        }

        // ---- 8 WMMAs per wave per K-step
        #pragma unroll
        for (int mi = 0; mi < 2; ++mi)
            #pragma unroll
            for (int ni = 0; ni < 4; ++ni)
                acc[mi][ni] = __builtin_amdgcn_wmma_f32_16x16x32_bf16(
                    false, aF[mi].v, false, bF[ni].v,
                    (short)0, acc[mi][ni], false, false);
        __syncthreads();
    }

    // ---- epilogue: C/D layout — lane holds N=lane%16, rows (lane/16)*8 + r
    const int mbase = (lane >> 4) * 8;
    #pragma unroll
    for (int mi = 0; mi < 2; ++mi)
        #pragma unroll
        for (int ni = 0; ni < 4; ++ni) {
            const int gm_b = m0 + waveM * 32 + mi * 16 + mbase;
            const int gp   = poff0 + waveN * 64 + ni * 16 + l16;
            #pragma unroll
            for (int r = 0; r < 8; ++r) {
                const int gm = gm_b + r;
                if (gm < M)
                    Y[((size_t)bimg * M + gm) * HW + gp] = acc[mi][ni][r];
            }
        }
}

// ---------------------------------------------------------------------------
// Weight packing: Wcat[896, Kpad] = [Wx | Wh(gates 0,1,2,6) | Wm(gates 3,4,5)]
// contraction order k = ci*25 + dy*5 + dx  (matches row-major filter layout)
// ---------------------------------------------------------------------------
__global__ void pack_wcat_kernel(const float* __restrict__ Wx,
                                 const float* __restrict__ Wh,
                                 const float* __restrict__ Wm,
                                 unsigned short* __restrict__ Wcat,
                                 int Cx, int Ktot, int Kpad)
{
    const size_t idx = (size_t)blockIdx.x * blockDim.x + threadIdx.x;
    const size_t total = (size_t)GATES7 * Kpad;
    if (idx >= total) return;
    const int rrow = (int)(idx / Kpad);
    const int k    = (int)(idx % Kpad);
    const int g    = rrow >> 7;       // gate 0..6
    const int j    = rrow & 127;
    float val = 0.0f;
    if (k < Ktot) {
        const int kx_end = Cx * 25;
        const int kh_end = kx_end + NH * 25;
        if (k < kx_end) {
            val = Wx[(size_t)rrow * kx_end + k];
        } else if (k < kh_end) {
            const int kh = k - kx_end;
            int hb = -1;
            if (g == 0) hb = 0; else if (g == 1) hb = 1;
            else if (g == 2) hb = 2; else if (g == 6) hb = 3;
            if (hb >= 0) val = Wh[((size_t)(hb * NH + j)) * (NH * 25) + kh];
        } else {
            const int km = k - kh_end;
            int mb = -1;
            if (g == 3) mb = 0; else if (g == 4) mb = 1; else if (g == 5) mb = 2;
            if (mb >= 0) val = Wm[((size_t)(mb * NH + j)) * (NH * 25) + km];
        }
    }
    Wcat[idx] = f2bf(val);
}

__global__ void f32_to_bf16_kernel(const float* __restrict__ src,
                                   unsigned short* __restrict__ dst, size_t n)
{
    const size_t i = (size_t)blockIdx.x * blockDim.x + threadIdx.x;
    if (i < n) dst[i] = f2bf(src[i]);
}

// net = mask*frame + (1-mask)*x_gen ; NHWC -> NCHW ; -> bf16
__global__ void net_kernel(const float* __restrict__ frames,
                           const float* __restrict__ mask_true,
                           const float* __restrict__ xg,
                           unsigned short* __restrict__ x_bf, int t)
{
    const int idx = blockIdx.x * blockDim.x + threadIdx.x;
    if (idx >= BATCH * FC * HW) return;
    const int b = idx / (FC * HW);
    const int r = idx - b * FC * HW;
    const int c = r / HW;
    const int p = r - c * HW;
    const float fr = frames[(((size_t)b * 20 + t) * HW + p) * FC + c];
    float mk = 1.0f;
    if (t >= T_PREV)
        mk = mask_true[(((size_t)b * (T_PREV - 1) + (t - T_PREV)) * HW + p) * FC + c];
    const float xv = xg[((size_t)b * FC + c) * HW + p];
    x_bf[idx] = f2bf(mk * fr + (1.0f - mk) * xv);
}

// xin = concat(x, h_prev, m_in) along channels (all bf16)
__global__ void pack_input_kernel(const unsigned short* __restrict__ x_src,
                                  const unsigned short* __restrict__ h_src,
                                  const unsigned short* __restrict__ cm_src,
                                  unsigned short* __restrict__ xin,
                                  int Cx, int Ccat)
{
    const size_t idx = (size_t)blockIdx.x * blockDim.x + threadIdx.x;
    if (idx >= (size_t)BATCH * Ccat * HW) return;
    const int b  = (int)(idx / ((size_t)Ccat * HW));
    const int r  = (int)(idx - (size_t)b * Ccat * HW);
    const int cc = r / HW;
    const int p  = r - cc * HW;
    unsigned short v;
    if (cc < Cx)            v = x_src[((size_t)b * Cx + cc) * HW + p];
    else if (cc < Cx + NH)  v = h_src[((size_t)b * NH + (cc - Cx)) * HW + p];
    else                    v = cm_src[((size_t)b * (2 * NH) + NH + (cc - Cx - NH)) * HW + p];
    xin[idx] = v;
}

// gates 0..5 -> c_new, m_new ; cm holds bf16(c_new)||bf16(m_new)
__global__ void gating1_kernel(const float* __restrict__ G,
                               float* __restrict__ c_st,
                               unsigned short* __restrict__ cm,
                               const unsigned short* __restrict__ cm_in)
{
    const int idx = blockIdx.x * blockDim.x + threadIdx.x;
    if (idx >= BATCH * NH * HW) return;
    const int b = idx / (NH * HW);
    const int r = idx - b * (NH * HW);
    const int j = r / HW;
    const int p = r - j * HW;
    const size_t gb = ((size_t)b * GATES7 + j) * HW + p;
    const float g0 = G[gb];
    const float g1 = G[gb + (size_t)1 * NH * HW];
    const float g2 = G[gb + (size_t)2 * NH * HW];
    const float g3 = G[gb + (size_t)3 * NH * HW];
    const float g4 = G[gb + (size_t)4 * NH * HW];
    const float g5 = G[gb + (size_t)5 * NH * HW];

    const size_t cidx = ((size_t)b * NH + j) * HW + p;
    const float c_old = c_st[cidx];
    const float m_in  = bf2f(cm_in[((size_t)b * (2 * NH) + NH + j) * HW + p]);

    const float i_t = sigm(g0);
    const float f_t = sigm(g1 + 1.0f);
    const float g_t = tanhf(g2);
    const float c_new = f_t * c_old + i_t * g_t;

    const float i_p = sigm(g3);
    const float f_p = sigm(g4 + 1.0f);
    const float g_p = tanhf(g5);
    const float m_new = f_p * m_in + i_p * g_p;

    c_st[cidx] = c_new;
    cm[((size_t)b * (2 * NH) + j) * HW + p]      = f2bf(c_new);
    cm[((size_t)b * (2 * NH) + NH + j) * HW + p] = f2bf(m_new);
}

// h = sigmoid(o_partial + conv_o) * tanh(conv_l)
__global__ void gating2_kernel(const float* __restrict__ G,
                               const float* __restrict__ S2o,
                               const float* __restrict__ S2l,
                               unsigned short* __restrict__ h_bf)
{
    const int idx = blockIdx.x * blockDim.x + threadIdx.x;
    if (idx >= BATCH * NH * HW) return;
    const int b = idx / (NH * HW);
    const int r = idx - b * (NH * HW);
    const int j = r / HW;
    const int p = r - j * HW;
    const float op = G[((size_t)b * GATES7 + 6 * NH + j) * HW + p];
    const size_t s = ((size_t)b * NH + j) * HW + p;
    const float o_t = sigm(op + S2o[s]);
    h_bf[s] = f2bf(o_t * tanhf(S2l[s]));
}

// x_gen [B,FC,HW] -> d_out [B,T,H,W,FC] at time t
__global__ void out_kernel(const float* __restrict__ xg,
                           float* __restrict__ out, int t)
{
    const int idx = blockIdx.x * blockDim.x + threadIdx.x;
    if (idx >= BATCH * FC * HW) return;
    const int b = idx / (FC * HW);
    const int r = idx - b * FC * HW;
    const int c = r / HW;
    const int p = r - c * HW;
    out[(((size_t)b * T_TOTAL + t) * HW + p) * FC + c] = xg[idx];
}

// ---------------------------------------------------------------------------
extern "C" void kernel_launch(void* const* d_in, const int* in_sizes, int n_in,
                              void* d_out, int out_size, void* d_ws, size_t ws_size,
                              hipStream_t stream)
{
    const float* frames = (const float*)d_in[0];
    const float* mask   = (const float*)d_in[1];
    const float* Wx0    = (const float*)d_in[2];
    const float* Wx     = (const float*)d_in[3];
    const float* Wh     = (const float*)d_in[4];
    const float* Wm     = (const float*)d_in[5];
    const float* Wo     = (const float*)d_in[6];
    const float* Wl     = (const float*)d_in[7];
    const float* Wout   = (const float*)d_in[8];
    float* outp = (float*)d_out;

    // ---- workspace carve-up (256B aligned) --------------------------------
    char* wp = (char*)d_ws;
    auto carve = [&](size_t bytes) -> void* {
        void* p = (void*)wp;
        wp += (bytes + 255) & ~(size_t)255;
        return p;
    };
    const int KPAD0 = 6816;   // (16+256)*25 = 6800 -> pad to %32
    const int KPADL = 9600;   // (128+256)*25
    unsigned short* wcat0 = (unsigned short*)carve((size_t)GATES7 * KPAD0 * 2);
    unsigned short* wcatL = (unsigned short*)carve((size_t)3 * GATES7 * KPADL * 2);
    unsigned short* wo_bf = (unsigned short*)carve((size_t)4 * NH * 6400 * 2);
    unsigned short* wl_bf = (unsigned short*)carve((size_t)4 * NH * 256 * 2);
    unsigned short* wout_bf = (unsigned short*)carve((size_t)FC * NH * 2);
    unsigned short* x_bf  = (unsigned short*)carve((size_t)BATCH * FC * HW * 2);
    unsigned short* h_bf  = (unsigned short*)carve((size_t)L_LAYERS * BATCH * NH * HW * 2);
    float*          c_st  = (float*)carve((size_t)L_LAYERS * BATCH * NH * HW * 4);
    unsigned short* cm_bf = (unsigned short*)carve((size_t)L_LAYERS * BATCH * 2 * NH * HW * 2);
    unsigned short* xin   = (unsigned short*)carve((size_t)BATCH * 384 * HW * 2);
    float*          G     = (float*)carve((size_t)BATCH * GATES7 * HW * 4);
    float*          S2o   = (float*)carve((size_t)BATCH * NH * HW * 4);
    float*          S2l   = (float*)carve((size_t)BATCH * NH * HW * 4);
    float*          xg    = (float*)carve((size_t)BATCH * FC * HW * 4);

    const int TB = 256;

    // ---- weight packing / conversion (once per call, deterministic) ------
    for (int l = 0; l < L_LAYERS; ++l) {
        const int Cx   = (l == 0) ? FC : NH;
        const int Ktot = (Cx + 2 * NH) * 25;
        const int Kpad = (l == 0) ? KPAD0 : KPADL;
        const float* wx_l = (l == 0) ? Wx0 : (Wx + (size_t)(l - 1) * GATES7 * NH * 25);
        const float* wh_l = Wh + (size_t)l * 4 * NH * NH * 25;
        const float* wm_l = Wm + (size_t)l * 3 * NH * NH * 25;
        unsigned short* wc = (l == 0) ? wcat0 : (wcatL + (size_t)(l - 1) * GATES7 * KPADL);
        const size_t n = (size_t)GATES7 * Kpad;
        pack_wcat_kernel<<<(unsigned)((n + TB - 1) / TB), TB, 0, stream>>>(
            wx_l, wh_l, wm_l, wc, Cx, Ktot, Kpad);
    }
    {
        size_t n = (size_t)4 * NH * 6400;
        f32_to_bf16_kernel<<<(unsigned)((n + TB - 1) / TB), TB, 0, stream>>>(Wo, wo_bf, n);
        n = (size_t)4 * NH * 256;
        f32_to_bf16_kernel<<<(unsigned)((n + TB - 1) / TB), TB, 0, stream>>>(Wl, wl_bf, n);
        n = (size_t)FC * NH;
        f32_to_bf16_kernel<<<(unsigned)((n + TB - 1) / TB), TB, 0, stream>>>(Wout, wout_bf, n);
    }

    // ---- zero recurrent state --------------------------------------------
    hipMemsetAsync(h_bf, 0, (size_t)L_LAYERS * BATCH * NH * HW * 2, stream);
    hipMemsetAsync(c_st, 0, (size_t)L_LAYERS * BATCH * NH * HW * 4, stream);
    hipMemsetAsync(cm_bf, 0, (size_t)L_LAYERS * BATCH * 2 * NH * HW * 2, stream);

    // ---- recurrence -------------------------------------------------------
    const int EW = BATCH * NH * HW;       // elementwise size per layer state
    for (int t = 0; t < T_TOTAL; ++t) {
        net_kernel<<<(BATCH * FC * HW + TB - 1) / TB, TB, 0, stream>>>(
            frames, mask, xg, x_bf, t);

        for (int l = 0; l < L_LAYERS; ++l) {
            const int Cx   = (l == 0) ? FC : NH;
            const int Ccat = Cx + 2 * NH;
            const int Ktot = Ccat * 25;
            const int Kpad = (l == 0) ? KPAD0 : KPADL;
            const unsigned short* wc  = (l == 0) ? wcat0
                                       : (wcatL + (size_t)(l - 1) * GATES7 * KPADL);
            const unsigned short* xsrc = (l == 0) ? x_bf
                                        : (h_bf + (size_t)(l - 1) * BATCH * NH * HW);
            unsigned short* h_l  = h_bf + (size_t)l * BATCH * NH * HW;
            float*          c_l  = c_st + (size_t)l * BATCH * NH * HW;
            unsigned short* cm_l = cm_bf + (size_t)l * BATCH * 2 * NH * HW;
            const unsigned short* cm_in = (l == 0)
                ? (cm_bf + (size_t)3 * BATCH * 2 * NH * HW)
                : (cm_bf + (size_t)(l - 1) * BATCH * 2 * NH * HW);

            const size_t npk = (size_t)BATCH * Ccat * HW;
            pack_input_kernel<<<(unsigned)((npk + TB - 1) / TB), TB, 0, stream>>>(
                xsrc, h_l, cm_in, xin, Cx, Ccat);

            // stage 1: all 7 gate pre-activations in one GEMM
            conv_gemm_wmma<<<dim3(GATES7 / 128, NPIX / 128), TB, 0, stream>>>(
                wc, xin, G, GATES7, Ccat, Ktot, Kpad, 25);

            gating1_kernel<<<(EW + TB - 1) / TB, TB, 0, stream>>>(G, c_l, cm_l, cm_in);

            // stage 2: conv(mem,Wo) 5x5 and conv(mem,Wl) 1x1
            conv_gemm_wmma<<<dim3(1, NPIX / 128), TB, 0, stream>>>(
                wo_bf + (size_t)l * NH * 6400, cm_l, S2o, NH, 2 * NH, 6400, 6400, 25);
            conv_gemm_wmma<<<dim3(1, NPIX / 128), TB, 0, stream>>>(
                wl_bf + (size_t)l * NH * 256, cm_l, S2l, NH, 2 * NH, 256, 256, 1);

            gating2_kernel<<<(EW + TB - 1) / TB, TB, 0, stream>>>(G, S2o, S2l, h_l);
        }

        // x_gen = 1x1 conv(h[3], Wout)
        conv_gemm_wmma<<<dim3(1, NPIX / 128), TB, 0, stream>>>(
            wout_bf, h_bf + (size_t)3 * BATCH * NH * HW, xg, FC, NH, NH, NH, 1);

        out_kernel<<<(BATCH * FC * HW + TB - 1) / TB, TB, 0, stream>>>(xg, outp, t);
    }
    (void)in_sizes; (void)n_in; (void)out_size; (void)ws_size;
}
